// PixelCNN_54468775248481
// MI455X (gfx1250) — matrix operand order repaired and checked by reading, not compiled, not run on hardware
//
#include <hip/hip_runtime.h>
#include <hip/hip_bf16.h>

typedef __attribute__((ext_vector_type(16))) _Float16 v16h;
typedef __attribute__((ext_vector_type(8)))  float    v8f;

#define NPIX 4096

// ---------------- WMMA fragment helpers (wave32, 16x16x32 f16) ----------------
// A-matrix: lane holds row M=lane&15; VGPR pair r -> K = (r<4 ? 2r : 2r+8) + (lane>=16 ? 8 : 0)
__device__ __forceinline__ v16h frag_rowK(const _Float16* p, int hi) {
  v16h f;
#pragma unroll
  for (int r = 0; r < 8; ++r) {
    int kb = ((r < 4) ? (2 * r) : (2 * r + 8)) + (hi ? 8 : 0);
    f[2 * r]     = p[kb];
    f[2 * r + 1] = p[kb + 1];
  }
  return f;
}
// B-matrix: lane holds column N=lane&15; VGPR pair r -> K = (r<4 ? 2r : 2r+8) + (lane>=16 ? 8 : 0)
__device__ __forceinline__ v16h frag_colK(const _Float16* p, int hi) {
  v16h f;
#pragma unroll
  for (int r = 0; r < 8; ++r) {
    int kb = ((r < 4) ? (2 * r) : (2 * r + 8)) + (hi ? 8 : 0);
    f[2 * r]     = p[kb];
    f[2 * r + 1] = p[kb + 1];
  }
  return f;
}

#define WMMA_F16(a, b, acc) \
  __builtin_amdgcn_wmma_f32_16x16x32_f16(false, (a), false, (b), (short)0, (acc), false, false)

// ---------------- direct kernels ----------------

// input conv: 3 -> 64, masked 5x5 (rows 0,1 full; row 2 cols 0,1)
__global__ void k_conv_in(const float* __restrict__ x, float* __restrict__ out,
                          const float* __restrict__ w, const float* __restrict__ bias) {
  int idx = blockIdx.x * blockDim.x + threadIdx.x;  // b*64*4096 + co*4096 + n
  int n = idx & 4095, co = (idx >> 12) & 63, b = idx >> 18;
  int y = n >> 6, x0 = n & 63;
  float acc = bias[co];
  for (int ci = 0; ci < 3; ++ci) {
    const float* xp = x + ((size_t)b * 3 + ci) * NPIX;
    const float* wp = w + ((size_t)co * 3 + ci) * 25;
    for (int tap = 0; tap < 12; ++tap) {
      int ky = tap < 5 ? 0 : (tap < 10 ? 1 : 2);
      int kx = tap - (ky == 0 ? 0 : (ky == 1 ? 5 : 10));
      int yy = y + ky - 2, xc = x0 + kx - 2;
      if (yy < 0 || xc < 0 || xc > 63) continue;
      acc += wp[ky * 5 + kx] * xp[yy * 64 + xc];
    }
  }
  out[idx] = acc;
}

// hidden conv: 64 -> 64, masked 5x5 (rows 0,1 full; row 2 cols 0,1,2 incl center)
__global__ void k_conv_cc(const float* __restrict__ h, float* __restrict__ out,
                          const float* __restrict__ w, const float* __restrict__ bias) {
  __shared__ float ws[64 * 13];
  int bc = blockIdx.y;  // b*64 + co
  int co = bc & 63, b = bc >> 6;
  for (int t = threadIdx.x; t < 64 * 13; t += blockDim.x) {
    int ci = t / 13, tap = t % 13;
    int ky = tap < 5 ? 0 : (tap < 10 ? 1 : 2);
    int kx = tap - (ky == 0 ? 0 : (ky == 1 ? 5 : 10));
    ws[t] = w[((size_t)(co * 64 + ci) * 5 + ky) * 5 + kx];
  }
  __syncthreads();
  int n = blockIdx.x * blockDim.x + threadIdx.x;
  int y = n >> 6, x = n & 63;
  float acc = bias[co];
  const float* hb = h + (size_t)b * 64 * NPIX;
  for (int tap = 0; tap < 13; ++tap) {
    int ky = tap < 5 ? 0 : (tap < 10 ? 1 : 2);
    int kx = tap - (ky == 0 ? 0 : (ky == 1 ? 5 : 10));
    int yy = y + ky - 2, xx = x + kx - 2;
    if (yy < 0 || xx < 0 || xx > 63) continue;
    int nn = yy * 64 + xx;
#pragma unroll 4
    for (int ci = 0; ci < 64; ++ci)
      acc += ws[ci * 13 + tap] * hb[(size_t)ci * NPIX + nn];
  }
  out[(size_t)bc * NPIX + n] = acc;
}

// per-channel batchnorm statistics over B*H*W = 8192 elements
__global__ void k_bn_stats(const float* __restrict__ src, float* __restrict__ mu,
                           float* __restrict__ rsig) {
  __shared__ float s1[256], s2[256];
  int c = blockIdx.x;
  float sum = 0.f, sq = 0.f;
  for (int i = threadIdx.x; i < 2 * NPIX; i += 256) {
    int b = i >> 12, n = i & 4095;
    float v = src[((size_t)b * 64 + c) * NPIX + n];
    sum += v; sq += v * v;
  }
  s1[threadIdx.x] = sum; s2[threadIdx.x] = sq;
  __syncthreads();
  for (int st = 128; st > 0; st >>= 1) {
    if (threadIdx.x < st) { s1[threadIdx.x] += s1[threadIdx.x + st]; s2[threadIdx.x] += s2[threadIdx.x + st]; }
    __syncthreads();
  }
  if (threadIdx.x == 0) {
    float m = s1[0] * (1.f / 8192.f);
    float var = s2[0] * (1.f / 8192.f) - m * m;
    mu[c] = m;
    rsig[c] = rsqrtf(var + 1e-5f);
  }
}

// bn apply + elu, optional skip save / skip add
__global__ void k_bn_apply(const float* __restrict__ src, float* __restrict__ dst,
                           const float* __restrict__ mu, const float* __restrict__ rsig,
                           const float* __restrict__ g, const float* __restrict__ bt,
                           const float* __restrict__ addskip, float* __restrict__ saveskip) {
  int idx = blockIdx.x * blockDim.x + threadIdx.x;  // over B*C*N
  int c = (idx >> 12) & 63;
  float v = g[c] * (src[idx] - mu[c]) * rsig[c] + bt[c];
  v = v > 0.f ? v : (expf(v) - 1.f);
  if (saveskip) saveskip[idx] = v;
  if (addskip) v += addskip[idx];
  dst[idx] = v;
}

// pack x0 = [h ; pos] into f16, token-major [b][n][96] (66 real channels, zero pad)
__global__ void k_pack_x0(const float* __restrict__ h, _Float16* __restrict__ x0) {
  int idx = blockIdx.x * blockDim.x + threadIdx.x;  // B*N*96
  int kc = idx % 96;
  int n = (idx / 96) & 4095;
  int b = idx / (96 * NPIX);
  float v;
  if (kc < 64)       v = h[((size_t)b * 64 + kc) * NPIX + n];
  else if (kc == 64) v = (float)(n >> 6) * (1.0f / 64.0f);
  else if (kc == 65) v = (float)(n & 63) * (1.0f / 64.0f);
  else               v = 0.f;
  x0[idx] = (_Float16)v;
}

// f32 -> f16 weight pack with K zero-padding, row-major [rows][dcol]
__global__ void k_pack_w(const float* __restrict__ src, _Float16* __restrict__ dst,
                         int rows, int scol, int dcol) {
  int idx = blockIdx.x * blockDim.x + threadIdx.x;
  if (idx >= rows * dcol) return;
  int r = idx / dcol, c = idx % dcol;
  dst[idx] = (c < scol) ? (_Float16)src[r * scol + c] : (_Float16)0.f;
}

// ---------------- WMMA GEMM kernels ----------------

// q/k projection: qT[n][kd] = relu(x0[n][:] . qw[kd][:] + qb), out padded to 32 cols
__global__ void k_proj_qk(const _Float16* __restrict__ x0, const _Float16* __restrict__ wq,
                          const float* __restrict__ qb, _Float16* __restrict__ q16) {
  int lane = threadIdx.x & 31, wave = threadIdx.x >> 5;
  int tile = blockIdx.x * 8 + wave;  // B * 256 n-tiles
  int b = tile >> 8, n0 = (tile & 255) << 4;
  int hi = lane >> 4, lr = lane & 15;
  const _Float16* A = x0 + ((size_t)b * NPIX + n0) * 96;
  v8f acc = {};
#pragma unroll
  for (int k0 = 0; k0 < 96; k0 += 32) {
    v16h a  = frag_rowK(A + (size_t)lr * 96 + k0, hi);
    v16h bf = frag_colK(wq + (size_t)lr * 96 + k0, hi);
    acc = WMMA_F16(a, bf, acc);
  }
  float bias = qb[lr];
  _Float16* outp = q16 + ((size_t)b * NPIX + n0) * 32;
#pragma unroll
  for (int r = 0; r < 8; ++r) {
    int m = r + hi * 8;
    float v = acc[r] + bias;
    v = v > 0.f ? v : 0.f;
    outp[(size_t)m * 32 + lr]      = (_Float16)v;
    outp[(size_t)m * 32 + lr + 16] = (_Float16)0.f;  // zero K-pad
  }
}

// v projection: wave computes 16 tokens x 32 channels (A fragments reused across
// 2 column tiles); stored transposed as [b][vc][m] for the a.v GEMM
__global__ void k_proj_v(const _Float16* __restrict__ x0, const _Float16* __restrict__ wv,
                         const float* __restrict__ vb, _Float16* __restrict__ v16o) {
  int lane = threadIdx.x & 31, wave = threadIdx.x >> 5;
  int tile = blockIdx.x * 8 + wave;  // B * 256 n-tiles * 2 col-groups
  int cg = tile & 1, nt = (tile >> 1) & 255, b = tile >> 9;
  int n0 = nt << 4;
  int hi = lane >> 4, lr = lane & 15;
  const _Float16* A = x0 + ((size_t)b * NPIX + n0 + lr) * 96;
  v16h af[3];
#pragma unroll
  for (int ks = 0; ks < 3; ++ks) af[ks] = frag_rowK(A + ks * 32, hi);
#pragma unroll
  for (int ct = 0; ct < 2; ++ct) {
    int vc0 = cg * 32 + ct * 16;
    const _Float16* Wp = wv + (size_t)(vc0 + lr) * 96;
    v8f acc = {};
#pragma unroll
    for (int ks = 0; ks < 3; ++ks) {
      v16h bf = frag_colK(Wp + ks * 32, hi);
      acc = WMMA_F16(af[ks], bf, acc);
    }
    int vc = vc0 + lr;
    float bias = vb[vc];
    _Float16* outp = v16o + ((size_t)b * 64 + vc) * NPIX + n0;
#pragma unroll
    for (int r = 0; r < 8; ++r) {
      float v = acc[r] + bias;
      v = v > 0.f ? v : 0.f;
      outp[r + hi * 8] = (_Float16)v;
    }
  }
}

// scores[n][m] = qT[n] . kT[m] (K=32, kd padded); wave covers 16 n x 64 m,
// q fragment loaded once; causal skip is wave-uniform (before any WMMA)
__global__ void k_scores(const _Float16* __restrict__ q16, const _Float16* __restrict__ k16,
                         _Float16* __restrict__ sc) {
  int lane = threadIdx.x & 31, wave = threadIdx.x >> 5;
  int grp = blockIdx.x * 8 + wave;  // 256 n-tiles * 64 m-groups
  int b = blockIdx.y;
  int nt = grp >> 6, mg = grp & 63;
  int n0 = nt << 4, m0 = mg << 6;
  if (m0 > n0) return;  // entire 64-wide group above the diagonal
  int hi = lane >> 4, lr = lane & 15;
  v16h a = frag_rowK(q16 + ((size_t)b * NPIX + n0 + lr) * 32, hi);
#pragma unroll
  for (int ct = 0; ct < 4; ++ct) {
    int m0c = m0 + ct * 16;
    if (m0c <= n0) {  // uniform per wave
      v16h bf = frag_colK(k16 + ((size_t)b * NPIX + m0c + lr) * 32, hi);
      v8f acc = {};
      acc = WMMA_F16(a, bf, acc);
      _Float16* outp = sc + ((size_t)b * NPIX + n0) * NPIX + m0c + lr;
#pragma unroll
      for (int r = 0; r < 8; ++r)
        outp[(size_t)(r + hi * 8) * NPIX] = (_Float16)acc[r];
    }
  }
}

// causal softmax, in-place on f16 scores; zeros the pad region (n, kend)
__global__ void k_softmax(_Float16* __restrict__ sc) {
  int row = blockIdx.x;  // b*4096 + n
  int n = row & 4095;
  _Float16* p = sc + (size_t)row * NPIX;
  __shared__ float red[256];
  int tid = threadIdx.x;
  int len = n + 1;
  float mx = -1e30f;
  for (int m = tid; m < len; m += 256) mx = fmaxf(mx, (float)p[m]);
  red[tid] = mx; __syncthreads();
  for (int st = 128; st > 0; st >>= 1) {
    if (tid < st) red[tid] = fmaxf(red[tid], red[tid + st]);
    __syncthreads();
  }
  mx = red[0]; __syncthreads();
  float sum = 0.f;
  for (int m = tid; m < len; m += 256) sum += expf((float)p[m] - mx);
  red[tid] = sum; __syncthreads();
  for (int st = 128; st > 0; st >>= 1) {
    if (tid < st) red[tid] += red[tid + st];
    __syncthreads();
  }
  float inv = 1.0f / red[0];
  int n0 = n & ~15;
  int kend = (n0 + 16 + 31) & ~31;
  if (kend > NPIX) kend = NPIX;
  for (int m = tid; m < kend; m += 256) {
    float v = (m < len) ? expf((float)p[m] - mx) * inv : 0.f;
    p[m] = (_Float16)v;
  }
}

// oT[n][vc] = sum_m a[n][m] * v[vc][m]; wave computes 16 tokens x 32 channels,
// A fragment shared by both accumulators; K-loop truncated causally
__global__ void k_attn_o(const _Float16* __restrict__ a16, const _Float16* __restrict__ v16i,
                         _Float16* __restrict__ o16) {
  int lane = threadIdx.x & 31, wave = threadIdx.x >> 5;
  int tile = blockIdx.x * 8 + wave;  // B * 256 n-tiles * 2 col-groups
  int cg = tile & 1, nt = (tile >> 1) & 255, b = tile >> 9;
  int n0 = nt << 4;
  int kend = (n0 + 16 + 31) & ~31;
  if (kend > NPIX) kend = NPIX;
  int hi = lane >> 4, lr = lane & 15;
  const _Float16* A  = a16 + ((size_t)b * NPIX + n0 + lr) * NPIX;
  const _Float16* B0 = v16i + ((size_t)b * 64 + cg * 32 + lr) * NPIX;
  const _Float16* B1 = B0 + (size_t)16 * NPIX;
  v8f acc0 = {}, acc1 = {};
#pragma unroll 2
  for (int k0 = 0; k0 < kend; k0 += 32) {
    __builtin_prefetch(A + k0 + 128, 0, 1);  // global_prefetch_b8, 256B ahead
    v16h a   = frag_rowK(A + k0, hi);
    v16h bf0 = frag_colK(B0 + k0, hi);
    v16h bf1 = frag_colK(B1 + k0, hi);
    acc0 = WMMA_F16(a, bf0, acc0);
    acc1 = WMMA_F16(a, bf1, acc1);
  }
  _Float16* outp = o16 + ((size_t)b * NPIX + n0) * 64 + cg * 32 + lr;
#pragma unroll
  for (int r = 0; r < 8; ++r) {
    outp[(size_t)(r + hi * 8) * 64]      = (_Float16)acc0[r];
    outp[(size_t)(r + hi * 8) * 64 + 16] = (_Float16)acc1[r];
  }
}

// final projection + residual: h[b][u][n] += relu(oT[n][:] . fw[u][:] + fb[u])
// wave computes 16 tokens x 32 channels, A fragments reused across column tiles
__global__ void k_attn_fw(const _Float16* __restrict__ o16, const _Float16* __restrict__ wf,
                          const float* __restrict__ fb, float* __restrict__ h) {
  int lane = threadIdx.x & 31, wave = threadIdx.x >> 5;
  int tile = blockIdx.x * 8 + wave;  // B * 256 * 2
  int cg = tile & 1, nt = (tile >> 1) & 255, b = tile >> 9;
  int n0 = nt << 4;
  int hi = lane >> 4, lr = lane & 15;
  const _Float16* A = o16 + ((size_t)b * NPIX + n0 + lr) * 64;
  v16h a0 = frag_rowK(A, hi);
  v16h a1 = frag_rowK(A + 32, hi);
#pragma unroll
  for (int ct = 0; ct < 2; ++ct) {
    int u0 = cg * 32 + ct * 16;
    const _Float16* Bw = wf + (size_t)(u0 + lr) * 64;
    v8f acc = {};
    acc = WMMA_F16(a0, frag_colK(Bw, hi), acc);
    acc = WMMA_F16(a1, frag_colK(Bw + 32, hi), acc);
    int u = u0 + lr;
    float bias = fb[u];
    float* hp = h + ((size_t)b * 64 + u) * NPIX + n0;
#pragma unroll
    for (int r = 0; r < 8; ++r) {
      float v = acc[r] + bias;
      v = v > 0.f ? v : 0.f;
      hp[r + hi * 8] += v;
    }
  }
}

// output head: sigmoid 1x1 conv -> reparam -> elu 1x1 -> sigmoid 1x1 (per pixel)
__global__ void k_head(const float* __restrict__ h, const float* __restrict__ noise,
                       const float* __restrict__ ow, const float* __restrict__ ob,
                       const float* __restrict__ f1w, const float* __restrict__ f1b,
                       const float* __restrict__ f2w, const float* __restrict__ f2b,
                       float* __restrict__ out) {
  int idx = blockIdx.x * blockDim.x + threadIdx.x;  // B*N
  int n = idx & 4095, b = idx >> 12;
  float hr[64];
#pragma unroll
  for (int c = 0; c < 64; ++c) hr[c] = h[((size_t)b * 64 + c) * NPIX + n];
  float s[50];
  for (int j = 0; j < 50; ++j) {
    float am = ob[j], al = ob[j + 50];
#pragma unroll
    for (int c = 0; c < 64; ++c) {
      am += ow[j * 64 + c] * hr[c];
      al += ow[(j + 50) * 64 + c] * hr[c];
    }
    float mean = 1.f / (1.f + expf(-am));
    float lv = (1.f / (1.f + expf(-al))) * 4.f - 2.f;
    s[j] = noise[((size_t)b * 50 + j) * NPIX + n] * expf(lv * 0.5f) + mean;
  }
  float s2[64];
  for (int u = 0; u < 64; ++u) {
    float acc = f1b[u];
#pragma unroll
    for (int j = 0; j < 50; ++j) acc += f1w[u * 50 + j] * s[j];
    s2[u] = acc > 0.f ? acc : (expf(acc) - 1.f);
  }
  for (int t = 0; t < 3; ++t) {
    float acc = f2b[t];
#pragma unroll
    for (int u = 0; u < 64; ++u) acc += f2w[t * 64 + u] * s2[u];
    out[((size_t)b * 3 + t) * NPIX + n] = 1.f / (1.f + expf(-acc));
  }
}

// ---------------- host ----------------
extern "C" void kernel_launch(void* const* d_in, const int* in_sizes, int n_in,
                              void* d_out, int out_size, void* d_ws, size_t ws_size,
                              hipStream_t stream) {
  const float* x       = (const float*)d_in[0];
  const float* noise   = (const float*)d_in[1];
  const float* in_w    = (const float*)d_in[2];
  const float* in_b    = (const float*)d_in[3];
  const float* in_g    = (const float*)d_in[4];
  const float* in_beta = (const float*)d_in[5];
  const float* cc_w    = (const float*)d_in[6];
  const float* cc_b    = (const float*)d_in[7];
  const float* qw      = (const float*)d_in[8];
  const float* qb      = (const float*)d_in[9];
  const float* kw      = (const float*)d_in[10];
  const float* kb      = (const float*)d_in[11];
  const float* vw      = (const float*)d_in[12];
  const float* vb      = (const float*)d_in[13];
  const float* fw      = (const float*)d_in[14];
  const float* fb      = (const float*)d_in[15];
  const float* bn_g    = (const float*)d_in[16];
  const float* bn_b    = (const float*)d_in[17];
  const float* out_w   = (const float*)d_in[18];
  const float* out_b   = (const float*)d_in[19];
  const float* f1w     = (const float*)d_in[20];
  const float* f1b     = (const float*)d_in[21];
  const float* f2w     = (const float*)d_in[22];
  const float* f2b     = (const float*)d_in[23];
  float* out = (float*)d_out;
  (void)in_sizes; (void)n_in; (void)out_size; (void)ws_size;

  char* ws = (char*)d_ws;
  size_t off = 0;
  auto alloc = [&](size_t bytes) -> void* {
    void* p = ws + off;
    off += (bytes + 255) & ~(size_t)255;
    return p;
  };
  float*     hA   = (float*)alloc(2ull * 64 * NPIX * 4);
  float*     hB   = (float*)alloc(2ull * 64 * NPIX * 4);
  float*     skip = (float*)alloc(2ull * 64 * NPIX * 4);
  float*     mu   = (float*)alloc(64 * 4);
  float*     rsig = (float*)alloc(64 * 4);
  _Float16*  x0   = (_Float16*)alloc(2ull * NPIX * 96 * 2);
  _Float16*  q16  = (_Float16*)alloc(2ull * NPIX * 32 * 2);
  _Float16*  k16  = (_Float16*)alloc(2ull * NPIX * 32 * 2);
  _Float16*  v16o = (_Float16*)alloc(2ull * 64 * NPIX * 2);
  _Float16*  o16  = (_Float16*)alloc(2ull * NPIX * 64 * 2);
  _Float16*  wq16 = (_Float16*)alloc(8ull * 16 * 96 * 2);
  _Float16*  wk16 = (_Float16*)alloc(8ull * 16 * 96 * 2);
  _Float16*  wv16 = (_Float16*)alloc(8ull * 64 * 96 * 2);
  _Float16*  wf16 = (_Float16*)alloc(8ull * 64 * 64 * 2);
  _Float16*  sc   = (_Float16*)alloc(2ull * NPIX * NPIX * 2);  // 64 MB

  // weight packing (all layers at once; per-layer slices are contiguous)
  k_pack_w<<<(8 * 16 * 96 + 255) / 256, 256, 0, stream>>>(qw, wq16, 8 * 16, 66, 96);
  k_pack_w<<<(8 * 16 * 96 + 255) / 256, 256, 0, stream>>>(kw, wk16, 8 * 16, 66, 96);
  k_pack_w<<<(8 * 64 * 96 + 255) / 256, 256, 0, stream>>>(vw, wv16, 8 * 64, 66, 96);
  k_pack_w<<<(8 * 64 * 64 + 255) / 256, 256, 0, stream>>>(fw, wf16, 8 * 64, 64, 64);

  // stem: masked conv + bn + elu
  k_conv_in<<<(2 * 64 * NPIX) / 256, 256, 0, stream>>>(x, hB, in_w, in_b);
  k_bn_stats<<<64, 256, 0, stream>>>(hB, mu, rsig);
  k_bn_apply<<<(2 * 64 * NPIX) / 256, 256, 0, stream>>>(hB, hA, mu, rsig, in_g, in_beta,
                                                        nullptr, nullptr);

  for (int i = 0; i < 8; ++i) {
    k_conv_cc<<<dim3(16, 128), 256, 0, stream>>>(hA, hB, cc_w + (size_t)i * 64 * 64 * 25,
                                                 cc_b + i * 64);
    k_pack_x0<<<(2 * NPIX * 96) / 256, 256, 0, stream>>>(hB, x0);
    k_proj_qk<<<64, 256, 0, stream>>>(x0, wq16 + (size_t)i * 16 * 96, qb + i * 16, q16);
    k_proj_qk<<<64, 256, 0, stream>>>(x0, wk16 + (size_t)i * 16 * 96, kb + i * 16, k16);
    k_proj_v<<<128, 256, 0, stream>>>(x0, wv16 + (size_t)i * 64 * 96, vb + i * 64, v16o);
    k_scores<<<dim3(2048, 2), 256, 0, stream>>>(q16, k16, sc);
    k_softmax<<<2 * NPIX, 256, 0, stream>>>(sc);
    k_attn_o<<<128, 256, 0, stream>>>(sc, v16o, o16);
    k_attn_fw<<<128, 256, 0, stream>>>(o16, wf16 + (size_t)i * 64 * 64, fb + i * 64, hB);
    k_bn_stats<<<64, 256, 0, stream>>>(hB, mu, rsig);
    k_bn_apply<<<(2 * 64 * NPIX) / 256, 256, 0, stream>>>(
        hB, hA, mu, rsig, bn_g + i * 64, bn_b + i * 64,
        (i >= 4) ? skip : nullptr, (i == 3) ? skip : nullptr);
  }

  k_head<<<(2 * NPIX) / 256, 256, 0, stream>>>(hA, noise, out_w, out_b, f1w, f1b, f2w, f2b, out);
}
